// SingleMambaBlock_15788299780382
// MI455X (gfx1250) — matrix operand order repaired and checked by reading, not compiled, not run on hardware
//
#include <hip/hip_runtime.h>
#include <hip/hip_bf16.h>
#include <math.h>

// ---------------- problem constants (fixed by the reference module) ---------
constexpr int Bb    = 2;
constexpr int Hs    = 64;
constexpr int Ws    = 64;
constexpr int Ltot  = Hs * Ws;          // 4096
constexpr int Dd    = 192;
constexpr int Nst   = 8;
constexpr int Rr    = 12;
constexpr int Kc    = 4;
constexpr int NDIR  = 4;
constexpr int MROWS = Bb * Ltot;        // 8192
constexpr int XD    = Rr + 2 * Nst;     // 28 (valid columns of x_dbl)
constexpr int XDP   = 64;               // padded x_dbl width (multiple of 64)

typedef __attribute__((ext_vector_type(16))) _Float16 v16h;
typedef __attribute__((ext_vector_type(8)))  _Float16 v8h;
typedef __attribute__((ext_vector_type(8)))  float    v8f;

// ---------------- direction permutations ------------------------------------
// reorder: u_re[l'] = u[perm(d,l')]   (reordered pos -> original pos)
__device__ __forceinline__ int permpos(int d, int l) {
  if (d == 0) return l;
  if (d == 1) return Ltot - 1 - l;
  int ll = (d == 3) ? (Ltot - 1 - l) : l;
  int w = ll / Hs, h = ll % Hs;
  return h * Ws + w;
}
// inv_reorder: y_sum[l] += y_d[qpos(d,l)]   (original pos -> reordered pos)
__device__ __forceinline__ int qpos(int d, int l) {
  if (d == 0) return l;
  if (d == 1) return Ltot - 1 - l;
  int h = l / Ws, w = l % Ws;
  int t = w * Hs + h;
  return (d == 3) ? (Ltot - 1 - t) : t;
}

__device__ __forceinline__ float siluf(float x) { return x / (1.f + __expf(-x)); }

// ---------------- f32 -> f16 weight convert ---------------------------------
__global__ void k_f32_to_f16(const float* __restrict__ in, _Float16* __restrict__ out, int n) {
  int i = blockIdx.x * blockDim.x + threadIdx.x;
  if (i < n) out[i] = (_Float16)in[i];
}

// Wx (NDIR,28,192) f32 -> (NDIR,64,192) f16, zero-padded rows 28..63
__global__ void k_pad_wx_f16(const float* __restrict__ Wx, _Float16* __restrict__ out) {
  int i = blockIdx.x * blockDim.x + threadIdx.x;
  int total = NDIR * XDP * Dd;
  if (i >= total) return;
  int k = i % Dd;
  int n = (i / Dd) % XDP;
  int d = i / (Dd * XDP);
  out[i] = (n < XD) ? (_Float16)Wx[((size_t)d * XD + n) * Dd + k] : (_Float16)0.f;
}

// ---------------- layernorm over D=192, one wave per row, emit f16 ----------
__global__ void k_ln_rows_f16(const float* __restrict__ X, const float* __restrict__ g,
                              const float* __restrict__ bta, _Float16* __restrict__ O,
                              int Mrows) {
  int lane = threadIdx.x & 31;
  int wave = threadIdx.x >> 5;
  int row  = blockIdx.x * (blockDim.x >> 5) + wave;
  if (row >= Mrows) return;
  const float* x = X + (size_t)row * Dd;
  float v[6]; float s = 0.f;
#pragma unroll
  for (int i = 0; i < 6; ++i) { v[i] = x[lane + 32 * i]; s += v[i]; }
#pragma unroll
  for (int o = 16; o > 0; o >>= 1) s += __shfl_xor(s, o, 32);
  float mu = s * (1.f / Dd);
  float q = 0.f;
#pragma unroll
  for (int i = 0; i < 6; ++i) { float dv = v[i] - mu; q += dv * dv; }
#pragma unroll
  for (int o = 16; o > 0; o >>= 1) q += __shfl_xor(q, o, 32);
  float rstd = rsqrtf(q * (1.f / Dd) + 1e-5f);
#pragma unroll
  for (int i = 0; i < 6; ++i) {
    int c = lane + 32 * i;
    O[(size_t)row * Dd + c] = (_Float16)((v[i] - mu) * rstd * g[c] + bta[c]);
  }
}

// ---------------- WMMA GEMM: C(MxN) = A(MxK,f16) * W(NxK,f16)^T [+ Res] -----
// Wave computes a 16x64 strip (4 accumulators), reusing the A fragment 4x.
// Requirements: M % 16 == 0, N % 64 == 0 per covered strip (all call sites
// use N in {384, 64, 192} with padded weights), K % 32 == 0.
__global__ void k_gemm_f16_wmma(const _Float16* __restrict__ A,
                                const _Float16* __restrict__ W,
                                float* __restrict__ C,
                                const float* __restrict__ Res,
                                int M, int N, int K) {
  int lane  = threadIdx.x & 31;
  int wave  = threadIdx.x >> 5;
  int tileM = blockIdx.x;
  int nslot = blockIdx.y * 4 + wave;          // 64 columns per slot
  if (nslot * 64 >= N) return;                // uniform per wave

  int mrow = tileM * 16 + (lane & 15);
  int kbA  = (lane >> 4) * 16;                // A frag K-base for lane half
  int kbB  = (lane >> 4) * 16;                // B frag K-base for lane half
  const _Float16* wp = W + ((size_t)nslot * 64 + (lane & 15)) * K;

  v8f acc[4] = {{}, {}, {}, {}};
  for (int k0 = 0; k0 < K; k0 += 32) {
    // A fragment: two contiguous 16B loads
    const _Float16* ap = A + (size_t)mrow * K + k0 + kbA;
    v8h alo = *(const v8h*)(ap);
    v8h ahi = *(const v8h*)(ap + 8);
    v16h afrag;
#pragma unroll
    for (int i = 0; i < 8; ++i) { afrag[i] = alo[i]; afrag[i + 8] = ahi[i]; }
    // 4 B fragments: one contiguous 32B load each
    v16h bfr[4];
#pragma unroll
    for (int j = 0; j < 4; ++j)
      bfr[j] = *(const v16h*)(wp + (size_t)(16 * j) * K + k0 + kbB);
#pragma unroll
    for (int j = 0; j < 4; ++j)
      acc[j] = __builtin_amdgcn_wmma_f32_16x16x32_f16(false, afrag, false, bfr[j],
                                                      (short)0, acc[j], false, false);
  }
#pragma unroll
  for (int j = 0; j < 4; ++j) {
    int n = nslot * 64 + j * 16 + (lane & 15);
#pragma unroll
    for (int v = 0; v < 8; ++v) {
      int m = tileM * 16 + v + 8 * (lane >> 4);
      size_t idx = (size_t)m * N + n;
      float r = Res ? Res[idx] : 0.f;
      C[idx] = acc[j][v] + r;
    }
  }
}

// ---------------- split xz -> xin(f16) + gate(silu, f32) --------------------
__global__ void k_split_silu(const float* __restrict__ xz, _Float16* __restrict__ xin_h,
                             float* __restrict__ gate) {
  size_t idx = blockIdx.x * (size_t)blockDim.x + threadIdx.x;
  if (idx >= (size_t)MROWS * Dd) return;
  int c = idx % Dd;
  size_t row = idx / Dd;
  float x = xz[row * (size_t)(2 * Dd) + c];
  float z = xz[row * (size_t)(2 * Dd) + Dd + c];
  xin_h[idx] = (_Float16)x;
  gate[idx]  = siluf(z);
}

// ---------------- per-direction causal depthwise conv + silu ----------------
__global__ void k_conv_dirs(const _Float16* __restrict__ xin,
                            const float* __restrict__ cw, const float* __restrict__ cb,
                            float* __restrict__ ucf, _Float16* __restrict__ uch) {
  size_t idx = blockIdx.x * (size_t)blockDim.x + threadIdx.x;
  size_t total = (size_t)NDIR * Bb * Ltot * Dd;
  if (idx >= total) return;
  int c = idx % Dd;
  int l = (idx / Dd) % Ltot;
  int b = (idx / ((size_t)Dd * Ltot)) % Bb;
  int d = idx / ((size_t)Dd * Ltot * Bb);
  float acc = cb[d * Dd + c];
#pragma unroll
  for (int k = 0; k < Kc; ++k) {
    int pos = l - (Kc - 1) + k;
    if (pos >= 0) {
      int src = permpos(d, pos);
      acc += cw[((size_t)d * Dd + c) * Kc + k] *
             (float)xin[((size_t)b * Ltot + src) * Dd + c];
    }
  }
  float s = siluf(acc);
  ucf[idx] = s;                 // layout (d,b,l,c) == idx
  uch[idx] = (_Float16)s;
}

// ---------------- selective scan: thread = (d,b,c,n); y reduced over n ------
__global__ void k_scan(const float* __restrict__ xdbl, const float* __restrict__ ucf,
                       const float* __restrict__ Wdt, const float* __restrict__ bdt,
                       const float* __restrict__ A_log, const float* __restrict__ Dp,
                       float* __restrict__ yd) {
  int t = blockIdx.x * blockDim.x + threadIdx.x;      // < NDIR*Bb*Dd*8
  int n = t & 7;
  int c = (t >> 3) % Dd;
  int b = ((t >> 3) / Dd) % Bb;
  int d = (t >> 3) / (Dd * Bb);
  if (d >= NDIR) return;

  float wdt[Rr];
#pragma unroll
  for (int r = 0; r < Rr; ++r) wdt[r] = Wdt[((size_t)d * Dd + c) * Rr + r];
  float bd = bdt[d * Dd + c];
  float Aa = -__expf(A_log[((size_t)d * Dd + c) * Nst + n]);
  float dp = Dp[d * Dd + c];

  const float* xrow = xdbl + ((size_t)d * MROWS + (size_t)b * Ltot) * XDP;
  const float* urow = ucf  + (((size_t)d * Bb + b) * (size_t)Ltot) * Dd;
  float*       yrow = yd   + (((size_t)d * Bb + b) * (size_t)Ltot) * Dd;

  float h = 0.f;
  for (int l = 0; l < Ltot; ++l) {
    const float* xr = xrow + (size_t)l * XDP;
    float s = bd;
#pragma unroll
    for (int r = 0; r < Rr; ++r) s += xr[r] * wdt[r];
    float delta = (s > 20.f) ? s : log1pf(__expf(s));
    float u  = urow[(size_t)l * Dd + c];
    float dA = __expf(delta * Aa);
    h = dA * h + delta * xr[Rr + n] * u;
    float yp = h * xr[Rr + Nst + n];
    // reduce over the 8 state lanes (contiguous in the wave)
    yp += __shfl_xor(yp, 1, 32);
    yp += __shfl_xor(yp, 2, 32);
    yp += __shfl_xor(yp, 4, 32);
    if (n == 0) yrow[(size_t)l * Dd + c] = yp + u * dp;
  }
}

// ---------------- inverse-reorder gather + gate mix -------------------------
__global__ void k_gather_mix(const float* __restrict__ yd, const float* __restrict__ gate,
                             float* __restrict__ ymix) {
  size_t idx = blockIdx.x * (size_t)blockDim.x + threadIdx.x;
  if (idx >= (size_t)MROWS * Dd) return;
  int c = idx % Dd;
  size_t rl = idx / Dd;
  int l = rl % Ltot;
  int b = rl / Ltot;
  float s = 0.f;
#pragma unroll
  for (int d = 0; d < NDIR; ++d) {
    int q = qpos(d, l);
    s += yd[(((size_t)d * Bb + b) * Ltot + q) * Dd + c];
  }
  ymix[idx] = s * 0.25f * gate[idx];
}

// ---------------- launcher ---------------------------------------------------
extern "C" void kernel_launch(void* const* d_in, const int* in_sizes, int n_in,
                              void* d_out, int out_size, void* d_ws, size_t ws_size,
                              hipStream_t stream) {
  const float* input  = (const float*)d_in[0];
  const float* ln_w   = (const float*)d_in[3];
  const float* ln_b   = (const float*)d_in[4];
  const float* W_in   = (const float*)d_in[5];
  const float* W_out  = (const float*)d_in[6];
  const float* conv_w = (const float*)d_in[7];
  const float* conv_b = (const float*)d_in[8];
  const float* Wx     = (const float*)d_in[9];
  const float* Wdt    = (const float*)d_in[10];
  const float* bdt    = (const float*)d_in[11];
  const float* A_log  = (const float*)d_in[12];
  const float* Dp     = (const float*)d_in[13];
  const float* yln_w  = (const float*)d_in[14];
  const float* yln_b  = (const float*)d_in[15];
  float* out = (float*)d_out;

  // bump allocator over workspace (256B aligned)
  char* p = (char*)d_ws;
  auto alloc = [&](size_t bytes) -> void* {
    void* r = (void*)p;
    p += (bytes + 255) & ~(size_t)255;
    return r;
  };
  _Float16* w_in_h  = (_Float16*)alloc((size_t)2 * Dd * Dd * sizeof(_Float16));       // 384x192
  _Float16* w_x_h   = (_Float16*)alloc((size_t)NDIR * XDP * Dd * sizeof(_Float16));   // 4x64x192 (padded)
  _Float16* w_out_h = (_Float16*)alloc((size_t)Dd * Dd * sizeof(_Float16));           // 192x192
  _Float16* xn_h    = (_Float16*)alloc((size_t)MROWS * Dd * sizeof(_Float16));
  float*    xz      = (float*)   alloc((size_t)MROWS * 2 * Dd * sizeof(float));
  _Float16* xin_h   = (_Float16*)alloc((size_t)MROWS * Dd * sizeof(_Float16));
  float*    gate    = (float*)   alloc((size_t)MROWS * Dd * sizeof(float));
  float*    uc_f    = (float*)   alloc((size_t)NDIR * MROWS * Dd * sizeof(float));
  _Float16* uc_h    = (_Float16*)alloc((size_t)NDIR * MROWS * Dd * sizeof(_Float16));
  float*    xdbl    = (float*)   alloc((size_t)NDIR * MROWS * XDP * sizeof(float));   // padded ld=64
  float*    ydbuf   = (float*)   alloc((size_t)NDIR * MROWS * Dd * sizeof(float));
  float*    ymix    = (float*)   alloc((size_t)MROWS * Dd * sizeof(float));
  _Float16* yg_h    = (_Float16*)alloc((size_t)MROWS * Dd * sizeof(_Float16));
  (void)ws_size;

  // 1) weight conversions to f16 (Wx zero-padded to 64 rows per direction)
  {
    int n1 = 2 * Dd * Dd, n2 = NDIR * XDP * Dd, n3 = Dd * Dd;
    k_f32_to_f16<<<(n1 + 255) / 256, 256, 0, stream>>>(W_in, w_in_h, n1);
    k_pad_wx_f16<<<(n2 + 255) / 256, 256, 0, stream>>>(Wx, w_x_h);
    k_f32_to_f16<<<(n3 + 255) / 256, 256, 0, stream>>>(W_out, w_out_h, n3);
  }
  // 2) layernorm(input) -> f16
  k_ln_rows_f16<<<MROWS / 8, 256, 0, stream>>>(input, ln_w, ln_b, xn_h, MROWS);
  // 3) xz = xn @ W_in^T   (M=8192, N=384, K=192)
  k_gemm_f16_wmma<<<dim3(MROWS / 16, (2 * Dd + 255) / 256), 128, 0, stream>>>(
      xn_h, w_in_h, xz, nullptr, MROWS, 2 * Dd, Dd);
  // 4) split + silu gate
  {
    size_t n = (size_t)MROWS * Dd;
    k_split_silu<<<(unsigned)((n + 255) / 256), 256, 0, stream>>>(xz, xin_h, gate);
  }
  // 5) 4-direction causal depthwise conv + silu
  {
    size_t n = (size_t)NDIR * MROWS * Dd;
    k_conv_dirs<<<(unsigned)((n + 255) / 256), 256, 0, stream>>>(xin_h, conv_w, conv_b,
                                                                 uc_f, uc_h);
  }
  // 6) x_dbl[d] = uc[d] @ WxPad[d]^T   (M=8192, N=64 padded, K=192)
  for (int d = 0; d < NDIR; ++d) {
    k_gemm_f16_wmma<<<dim3(MROWS / 16, 1), 128, 0, stream>>>(
        uc_h + (size_t)d * MROWS * Dd, w_x_h + (size_t)d * XDP * Dd,
        xdbl + (size_t)d * MROWS * XDP, nullptr, MROWS, XDP, Dd);
  }
  // 7) selective scan (thread per (dir,b,channel,state))
  {
    int nthr = NDIR * Bb * Dd * Nst;  // 12288
    k_scan<<<nthr / 256, 256, 0, stream>>>(xdbl, uc_f, Wdt, bdt, A_log, Dp, ydbuf);
  }
  // 8) inverse reorder gather + gate
  {
    size_t n = (size_t)MROWS * Dd;
    k_gather_mix<<<(unsigned)((n + 255) / 256), 256, 0, stream>>>(ydbuf, gate, ymix);
  }
  // 9) y layernorm -> f16
  k_ln_rows_f16<<<MROWS / 8, 256, 0, stream>>>(ymix, yln_w, yln_b, yg_h, MROWS);
  // 10) out = yg @ W_out^T + input   (M=8192, N=192, K=192)
  k_gemm_f16_wmma<<<dim3(MROWS / 16, (Dd + 255) / 256), 128, 0, stream>>>(
      yg_h, w_out_h, out, input, MROWS, Dd, Dd);
}
